// SocialPoolingLayer_14448269984518
// MI455X (gfx1250) — compile-verified, matrix operand order checked by reading, not run on hardware
//
#include <hip/hip_runtime.h>
#include <hip/hip_bf16.h>

// ---------------------------------------------------------------------------
// SocialPoolingLayer: per-edge 3-matmul MLP + sigmoid gate + scatter-mean.
// 16-edge tiles per wave, v_wmma_f32_16x16x32_bf16, f32 accumulation,
// hardware v_cvt_pk_bf16_f32 conversions, LDS bf16 weights, f32 staging,
// atomic scatter + finalize.
// ---------------------------------------------------------------------------

typedef __attribute__((ext_vector_type(16))) __bf16 bf16x16;
typedef __attribute__((ext_vector_type(2)))  __bf16 bf16x2;
typedef __attribute__((ext_vector_type(8)))  float  f32x8;
typedef __attribute__((ext_vector_type(2)))  float  f32x2;

#define DD        64
#define TWOD      128
#define TILE      16
#define WPB       8            // waves per block
#define BLOCK     (WPB * 32)

union frag_u { unsigned u[8]; bf16x16 f; };

// pack two fp32 into one dword of bf16x2 via hardware packed convert
__device__ __forceinline__ unsigned pack_bf16(float x, float y) {
    f32x2 v = {x, y};
    bf16x2 b = __builtin_convertvector(v, bf16x2);
    return __builtin_bit_cast(unsigned, b);
}

__global__ void __launch_bounds__(BLOCK)
edge_mlp_wmma_kernel(const float* __restrict__ node_emb,
                     const long long* __restrict__ eidx,
                     const float* __restrict__ W1, const float* __restrict__ b1,
                     const float* __restrict__ W2, const float* __restrict__ b2,
                     const float* __restrict__ Wg, const float* __restrict__ bg,
                     float* __restrict__ summed, float* __restrict__ counts,
                     long long E)
{
    // weights transposed ([n][k]) as bf16 so B fragments are contiguous dwords
    __shared__ unsigned sW1[DD * TWOD / 2];        // 16KB
    __shared__ unsigned sW2[DD * DD / 2];          // 8KB
    __shared__ unsigned sWg[DD * DD / 2];          // 8KB
    __shared__ float    sb1[DD], sb2[DD], sbg[DD];
    __shared__ float    sStage[WPB][TILE * DD];    // per-wave f32 staging, 32KB

    // ---- cooperative weight conversion (fp32 -> bf16 pairs, transposed) ----
    for (int i = threadIdx.x; i < DD * TWOD / 2; i += BLOCK) {
        int n = i / (TWOD / 2), kp = i % (TWOD / 2);
        int k = kp * 2;
        sW1[i] = pack_bf16(W1[k * DD + n], W1[(k + 1) * DD + n]);
    }
    for (int i = threadIdx.x; i < DD * DD / 2; i += BLOCK) {
        int n = i / (DD / 2), kp = i % (DD / 2);
        int k = kp * 2;
        sW2[i] = pack_bf16(W2[k * DD + n], W2[(k + 1) * DD + n]);
        sWg[i] = pack_bf16(Wg[k * DD + n], Wg[(k + 1) * DD + n]);
    }
    for (int i = threadIdx.x; i < DD; i += BLOCK) {
        sb1[i] = b1[i]; sb2[i] = b2[i]; sbg[i] = bg[i];
    }
    __syncthreads();

    const int wave = threadIdx.x >> 5;
    const int lane = threadIdx.x & 31;
    const int hi   = lane >> 4;     // half-wave select
    const int ln   = lane & 15;     // row (A) or column (B/C) id
    float* stage = sStage[wave];

    const long long numTiles = (E + TILE - 1) / TILE;
    for (long long tile = (long long)blockIdx.x * WPB + wave; tile < numTiles;
         tile += (long long)gridDim.x * WPB)
    {
        const long long e = tile * TILE + ln;     // edge row owned by this lane
        const bool valid = (e < E);
        const int srcN = valid ? (int)eidx[e] : 0;
        const int dstN = valid ? (int)eidx[E + e] : 0;

        // ---------------- Pass 1: H = relu(pair_emb @ W1 + b1) ----------------
        f32x8 acc[4] = {};
        #pragma unroll
        for (int kt = 0; kt < 4; ++kt) {
            // gather A fragment (16x32 bf16): this half-wave needs K = hi*8+{0..7}
            // and hi*8+{16..23} of the 32-wide tile -> 4 aligned float4 loads
            const int node = (kt < 2) ? srcN : dstN;
            const float* rowp = node_emb + (long long)node * DD + (kt & 1) * 32 + hi * 8;
            float4 f0 = ((const float4*)rowp)[0];
            float4 f1 = ((const float4*)rowp)[1];
            float4 f2 = ((const float4*)(rowp + 16))[0];
            float4 f3 = ((const float4*)(rowp + 16))[1];
            frag_u a;
            a.u[0] = pack_bf16(f0.x, f0.y);  a.u[1] = pack_bf16(f0.z, f0.w);
            a.u[2] = pack_bf16(f1.x, f1.y);  a.u[3] = pack_bf16(f1.z, f1.w);
            a.u[4] = pack_bf16(f2.x, f2.y);  a.u[5] = pack_bf16(f2.z, f2.w);
            a.u[6] = pack_bf16(f3.x, f3.y);  a.u[7] = pack_bf16(f3.z, f3.w);

            #pragma unroll
            for (int nt = 0; nt < 4; ++nt) {
                // B fragment: column n = nt*16+ln, dwords hi*4+{0..3} and +8
                const unsigned* wp = sW1 + (nt * 16 + ln) * (TWOD / 2)
                                        + kt * 16 + hi * 4;
                frag_u b;
                #pragma unroll
                for (int v = 0; v < 4; ++v) { b.u[v] = wp[v]; b.u[v + 4] = wp[v + 8]; }
                acc[nt] = __builtin_amdgcn_wmma_f32_16x16x32_bf16(
                    false, a.f, false, b.f, (short)0, acc[nt], false, false);
            }
        }
        // bias + relu, stage H (f32) through LDS to re-form A layout
        #pragma unroll
        for (int nt = 0; nt < 4; ++nt) {
            const int col = nt * 16 + ln;
            const float bias = sb1[col];
            #pragma unroll
            for (int v = 0; v < 8; ++v) {
                float h = acc[nt][v] + bias;
                stage[(v + hi * 8) * DD + col] = h > 0.f ? h : 0.f;
            }
        }
        asm volatile("s_wait_dscnt 0" ::: "memory");
        __builtin_amdgcn_wave_barrier();

        // ---------------- Pass 2: I = H @ W2 + b2 ----------------
        f32x8 acc2[4] = {};
        #pragma unroll
        for (int kt = 0; kt < 2; ++kt) {
            const float* sp = stage + ln * DD + kt * 32 + hi * 8;
            float4 f0 = ((const float4*)sp)[0];
            float4 f1 = ((const float4*)sp)[1];
            float4 f2 = ((const float4*)(sp + 16))[0];
            float4 f3 = ((const float4*)(sp + 16))[1];
            frag_u a;
            a.u[0] = pack_bf16(f0.x, f0.y);  a.u[1] = pack_bf16(f0.z, f0.w);
            a.u[2] = pack_bf16(f1.x, f1.y);  a.u[3] = pack_bf16(f1.z, f1.w);
            a.u[4] = pack_bf16(f2.x, f2.y);  a.u[5] = pack_bf16(f2.z, f2.w);
            a.u[6] = pack_bf16(f3.x, f3.y);  a.u[7] = pack_bf16(f3.z, f3.w);

            #pragma unroll
            for (int nt = 0; nt < 4; ++nt) {
                const unsigned* wp = sW2 + (nt * 16 + ln) * (DD / 2)
                                        + kt * 16 + hi * 4;
                frag_u b;
                #pragma unroll
                for (int v = 0; v < 4; ++v) { b.u[v] = wp[v]; b.u[v + 4] = wp[v + 8]; }
                acc2[nt] = __builtin_amdgcn_wmma_f32_16x16x32_bf16(
                    false, a.f, false, b.f, (short)0, acc2[nt], false, false);
            }
        }
        __builtin_amdgcn_wave_barrier();   // order LDS reads above vs writes below
        #pragma unroll
        for (int nt = 0; nt < 4; ++nt) {
            const int col = nt * 16 + ln;
            const float bias = sb2[col];
            #pragma unroll
            for (int v = 0; v < 8; ++v) {
                float x = acc2[nt][v] + bias;
                acc2[nt][v] = x;                       // keep f32 interaction
                stage[(v + hi * 8) * DD + col] = x;
            }
        }
        asm volatile("s_wait_dscnt 0" ::: "memory");
        __builtin_amdgcn_wave_barrier();

        // ---------------- Pass 3: G = I @ Wg + bg ----------------
        f32x8 acc3[4] = {};
        #pragma unroll
        for (int kt = 0; kt < 2; ++kt) {
            const float* sp = stage + ln * DD + kt * 32 + hi * 8;
            float4 f0 = ((const float4*)sp)[0];
            float4 f1 = ((const float4*)sp)[1];
            float4 f2 = ((const float4*)(sp + 16))[0];
            float4 f3 = ((const float4*)(sp + 16))[1];
            frag_u a;
            a.u[0] = pack_bf16(f0.x, f0.y);  a.u[1] = pack_bf16(f0.z, f0.w);
            a.u[2] = pack_bf16(f1.x, f1.y);  a.u[3] = pack_bf16(f1.z, f1.w);
            a.u[4] = pack_bf16(f2.x, f2.y);  a.u[5] = pack_bf16(f2.z, f2.w);
            a.u[6] = pack_bf16(f3.x, f3.y);  a.u[7] = pack_bf16(f3.z, f3.w);

            #pragma unroll
            for (int nt = 0; nt < 4; ++nt) {
                const unsigned* wp = sWg + (nt * 16 + ln) * (DD / 2)
                                        + kt * 16 + hi * 4;
                frag_u b;
                #pragma unroll
                for (int v = 0; v < 4; ++v) { b.u[v] = wp[v]; b.u[v + 4] = wp[v + 8]; }
                acc3[nt] = __builtin_amdgcn_wmma_f32_16x16x32_bf16(
                    false, a.f, false, b.f, (short)0, acc3[nt], false, false);
            }
        }

        // ------------- gated = I * sigmoid(G); scatter-add to src node -------------
        int  rowsrc[8];
        bool rowok[8];
        #pragma unroll
        for (int v = 0; v < 8; ++v) {
            const int row = v + hi * 8;
            rowsrc[v] = __shfl(srcN, row, 32);   // lanes 0..15 hold row's src node
            rowok[v]  = (tile * TILE + row) < E;
        }
        #pragma unroll
        for (int nt = 0; nt < 4; ++nt) {
            const int col = nt * 16 + ln;
            const float bias = sbg[col];
            #pragma unroll
            for (int v = 0; v < 8; ++v) {
                float g = 1.f / (1.f + __expf(-(acc3[nt][v] + bias)));
                float out = acc2[nt][v] * g;
                if (rowok[v])
                    atomicAdd(&summed[(long long)rowsrc[v] * DD + col], out);
            }
        }
        if (hi == 0 && valid)
            atomicAdd(&counts[srcN], 1.0f);
    }
}

__global__ void zero_kernel(float* __restrict__ p, long long n) {
    long long i = (long long)blockIdx.x * blockDim.x + threadIdx.x;
    if (i < n) p[i] = 0.f;
}

__global__ void finalize_kernel(const float* __restrict__ summed,
                                const float* __restrict__ counts,
                                float* __restrict__ out, long long total) {
    long long i = (long long)blockIdx.x * blockDim.x + threadIdx.x;
    if (i < total) {
        float c = counts[i >> 6];               // DD == 64
        out[i] = summed[i] / fmaxf(c, 1.0f);
    }
}

extern "C" void kernel_launch(void* const* d_in, const int* in_sizes, int n_in,
                              void* d_out, int out_size, void* d_ws, size_t ws_size,
                              hipStream_t stream) {
    const float*     node_emb = (const float*)d_in[0];
    const long long* eidx     = (const long long*)d_in[1];   // int64 [2, E]
    const float*     W1 = (const float*)d_in[2];
    const float*     b1 = (const float*)d_in[3];
    const float*     W2 = (const float*)d_in[4];
    const float*     b2 = (const float*)d_in[5];
    const float*     Wg = (const float*)d_in[6];
    const float*     bg = (const float*)d_in[7];
    float* out = (float*)d_out;

    const long long E = (long long)in_sizes[1] / 2;
    const long long N = (long long)out_size / DD;

    float* summed = (float*)d_ws;               // N*64 floats
    float* counts = summed + (size_t)N * DD;    // N floats

    // zero the workspace accumulators (d_ws is poisoned each run)
    const long long zn = N * DD + N;
    zero_kernel<<<dim3((unsigned)((zn + 255) / 256)), dim3(256), 0, stream>>>(summed, zn);

    const long long tiles = (E + TILE - 1) / TILE;
    unsigned blocks = (unsigned)((tiles + WPB - 1) / WPB);
    edge_mlp_wmma_kernel<<<dim3(blocks), dim3(BLOCK), 0, stream>>>(
        node_emb, eidx, W1, b1, W2, b2, Wg, bg, summed, counts, E);

    const long long total = N * DD;
    finalize_kernel<<<dim3((unsigned)((total + 255) / 256)), dim3(256), 0, stream>>>(
        summed, counts, out, total);
}
